// Graph_GAN_70635032150621
// MI455X (gfx1250) — matrix operand order repaired and checked by reading, not compiled, not run on hardware
//
#include <hip/hip_runtime.h>

// ---------------- problem constants ----------------
#define ALPHA_SLOPE 0.2f
static constexpr int Bq   = 32;    // batch
static constexpr int Nq   = 100;   // nodes
static constexpr int NPAD = 112;   // nodes padded to 7*16
static constexpr int NT   = 7;     // 16-row tiles per batch

typedef __attribute__((ext_vector_type(16))) _Float16 h16;
typedef __attribute__((ext_vector_type(8)))  _Float16 h8;
typedef __attribute__((ext_vector_type(8)))  float    f8;

// Edge-kernel LDS: W1t[160][104] + W2t[192][168] + a0[256][104] + a1[256][168] (f16)
//                  + agg[16][192] (f32)  = 249344 bytes  (< 320KB WGP LDS)
static constexpr int EDGE_LDS = (160*104 + 192*168 + 256*104 + 256*168) * 2 + 16*192*4;
// Node-kernel LDS: 8 waves * 2 slabs * [16][264] f16 = 135168 bytes
static constexpr int NODE_LDS = 8 * 2 * 16 * 264 * 2;

// lrelu(x) = max(x, 0.2*x) for alpha in (0,1): 2 VALU ops, no cmp/cndmask
__device__ __forceinline__ float lrelu(float x) { return fmaxf(x, ALPHA_SLOPE * x); }

// Build a 16x32 f16 WMMA fragment from two contiguous 16B LDS/global chunks.
__device__ __forceinline__ h16 ld_frag(const _Float16* p0, const _Float16* p1)
{
    h8 lo = *(const h8*)p0;
    h8 hi = *(const h8*)p1;
    h16 r;
#pragma unroll
    for (int i = 0; i < 8; ++i) { r[i] = lo[i]; r[i + 8] = hi[i]; }
    return r;
}

#define WMMA_F16(Af, Bf, Cc) \
    __builtin_amdgcn_wmma_f32_16x16x32_f16(false, (Af), false, (Bf), (short)0, (Cc), false, false)

// ---------------------------------------------------------------------------
// Prep kernels
// ---------------------------------------------------------------------------
__global__ void init_x_kernel(const float* __restrict__ x, float* __restrict__ xA,
                              float* __restrict__ xB)
{
    int idx = blockIdx.x * 256 + threadIdx.x;          // Bq*NPAD*32
    if (idx >= Bq * NPAD * 32) return;
    int k = idx % 32, n = (idx / 32) % NPAD, b = idx / (32 * NPAD);
    xA[idx] = (n < Nq) ? x[(size_t)(b * Nq + n) * 32 + k] : 0.f;
    xB[idx] = 0.f;
}

// dst[n*K + k] = (f16) src[k*Nn + n]   (W -> W^T, f32 -> f16)
__global__ void transpose_cvt_kernel(_Float16* __restrict__ dst, const float* __restrict__ src,
                                     int K, int Nn)
{
    int idx = blockIdx.x * 256 + threadIdx.x;
    if (idx >= K * Nn) return;
    int n = idx / K, k = idx % K;
    dst[idx] = (_Float16)src[(size_t)k * Nn + n];
}

// P[b,n,:] = x[b,n] @ W0[0:32,:] ;  Q[b,n,:] = x[b,n] @ W0[32:64,:] + b0
__global__ __launch_bounds__(256) void pq_kernel(const float* __restrict__ x,
                                                 const float* __restrict__ w0,
                                                 const float* __restrict__ b0,
                                                 float* __restrict__ P, float* __restrict__ Q)
{
    int idx = blockIdx.x * 256 + threadIdx.x;          // Bq*NPAD*96
    if (idx >= Bq * NPAD * 96) return;
    int o = idx % 96;
    const float* xr = x + (size_t)(idx / 96) * 32;
    float p = 0.f, q = b0[o];
#pragma unroll
    for (int k = 0; k < 32; ++k) {
        float xv = xr[k];
        p += xv * w0[k * 96 + o];
        q += xv * w0[(32 + k) * 96 + o];
    }
    P[idx] = p; Q[idx] = q;
}

// ---------------------------------------------------------------------------
// Edge MLP + neighbor aggregation. One workgroup per (batch, i-tile).
// Each wave owns M-tiles {2w, 2w+1}; A-fragments are register-resident across
// the full N loop; each B-fragment feeds two independent WMMAs.
// ---------------------------------------------------------------------------
__global__ __launch_bounds__(256)
void edge_kernel(const float* __restrict__ P, const float* __restrict__ Q,
                 const _Float16* __restrict__ w1t, const _Float16* __restrict__ w2t,
                 const float* __restrict__ b1, const float* __restrict__ b2,
                 float* __restrict__ aggG)
{
    extern __shared__ char smem[];
    _Float16* sW1 = (_Float16*)smem;          // [160][104]
    _Float16* sW2 = sW1 + 160 * 104;          // [192][168]
    _Float16* sA0 = sW2 + 192 * 168;          // [256][104]  edges x 96
    _Float16* sA1 = sA0 + 256 * 104;          // [256][168]  edges x 160
    float*   sAgg = (float*)(sA1 + 256 * 168);// [16][192]

    const int tid  = threadIdx.x;
    const int wave = tid >> 5, lane = tid & 31;
    const int b    = blockIdx.x / NT;
    const int it   = blockIdx.x % NT;
    const int i0   = it * 16;
    const int lr   = lane & 15;
    const int hi16 = (lane & 16) ? 1 : 0;
    const int m0   = wave * 2;                // this wave's two M-tiles

    // cooperative weight staging (rows padded to odd multiples of 16B)
    for (int id = tid; id < 160 * 12; id += 256) {
        int r = id / 12, c = (id % 12) * 8;
        *(h8*)(sW1 + r * 104 + c) = *(const h8*)(w1t + r * 96 + c);
    }
    for (int id = tid; id < 192 * 20; id += 256) {
        int r = id / 20, c = (id % 20) * 8;
        *(h8*)(sW2 + r * 168 + c) = *(const h8*)(w2t + r * 160 + c);
    }
    for (int id = tid; id < 16 * 192; id += 256) sAgg[id] = 0.f;
    __syncthreads();

    for (int jt = 0; jt < NT; ++jt) {
        const int j0 = jt * 16;

        // a0[e][k] = lrelu(P[i0+e/16][k] + Q[j0+e%16][k]); edge e = il*16 + jl
        {
            const int e  = tid;
            const int il = e >> 4, jl = e & 15;
            const float4* Pr = (const float4*)(P + (size_t)(b * NPAD + i0 + il) * 96);
            const float4* Qr = (const float4*)(Q + (size_t)(b * NPAD + j0 + jl) * 96);
            _Float16* dst = sA0 + e * 104;
#pragma unroll
            for (int c = 0; c < 12; ++c) {      // 12 x (8 floats -> 8 halves)
                float4 pa = Pr[2 * c], pb = Pr[2 * c + 1];
                float4 qa = Qr[2 * c], qb = Qr[2 * c + 1];
                h8 o;
                o[0] = (_Float16)lrelu(pa.x + qa.x);
                o[1] = (_Float16)lrelu(pa.y + qa.y);
                o[2] = (_Float16)lrelu(pa.z + qa.z);
                o[3] = (_Float16)lrelu(pa.w + qa.w);
                o[4] = (_Float16)lrelu(pb.x + qb.x);
                o[5] = (_Float16)lrelu(pb.y + qb.y);
                o[6] = (_Float16)lrelu(pb.z + qb.z);
                o[7] = (_Float16)lrelu(pb.w + qb.w);
                *(h8*)(dst + 8 * c) = o;
            }
        }
        __syncthreads();

        // ---- L1: a1 = lrelu(a0 @ W1 + b1)  [256 x 160] ----
        {
            h16 Af[2][3];
#pragma unroll
            for (int mm = 0; mm < 2; ++mm)
#pragma unroll
                for (int kc = 0; kc < 3; ++kc) {
                    const _Float16* ap = sA0 + ((m0 + mm) * 16 + lr) * 104 + kc * 32 + (hi16 ? 8 : 0);
                    Af[mm][kc] = ld_frag(ap, ap + 16);
                }
            for (int n1 = 0; n1 < 10; ++n1) {
                f8 acc0 = {}, acc1 = {};
#pragma unroll
                for (int kc = 0; kc < 3; ++kc) {
                    const _Float16* bp = sW1 + (n1 * 16 + lr) * 104 + kc * 32 + (hi16 ? 8 : 0);
                    h16 bm = ld_frag(bp, bp + 16);
                    acc0 = WMMA_F16(Af[0][kc], bm, acc0);
                    acc1 = WMMA_F16(Af[1][kc], bm, acc1);
                }
                const int col = n1 * 16 + lr;
                const float bias = b1[col];
#pragma unroll
                for (int v = 0; v < 8; ++v) {
                    sA1[((m0 + 0) * 16 + v + 8 * hi16) * 168 + col] = (_Float16)lrelu(acc0[v] + bias);
                    sA1[((m0 + 1) * 16 + v + 8 * hi16) * 168 + col] = (_Float16)lrelu(acc1[v] + bias);
                }
            }
        }
        __syncthreads();

        // ---- L2: a2 = lrelu(a1 @ W2 + b2) [256 x 192]; masked row-sum into sAgg ----
        {
            h16 Af[2][5];
#pragma unroll
            for (int mm = 0; mm < 2; ++mm)
#pragma unroll
                for (int kc = 0; kc < 5; ++kc) {
                    const _Float16* ap = sA1 + ((m0 + mm) * 16 + lr) * 168 + kc * 32 + (hi16 ? 8 : 0);
                    Af[mm][kc] = ld_frag(ap, ap + 16);
                }
            for (int n2 = 0; n2 < 12; ++n2) {
                f8 acc0 = {}, acc1 = {};
#pragma unroll
                for (int kc = 0; kc < 5; ++kc) {
                    const _Float16* bp = sW2 + (n2 * 16 + lr) * 168 + kc * 32 + (hi16 ? 8 : 0);
                    h16 bm = ld_frag(bp, bp + 16);
                    acc0 = WMMA_F16(Af[0][kc], bm, acc0);
                    acc1 = WMMA_F16(Af[1][kc], bm, acc1);
                }
                const int col = n2 * 16 + lr;
                const float bias = b2[col];
                float s0 = 0.f, s1 = 0.f;
#pragma unroll
                for (int v = 0; v < 8; ++v) {
                    int jl = v + 8 * hi16;                 // tile rows are j_local
                    float valid = (j0 + jl < Nq) ? 1.f : 0.f;
                    s0 += valid * lrelu(acc0[v] + bias);
                    s1 += valid * lrelu(acc1[v] + bias);
                }
                s0 += __shfl_xor(s0, 16, 32);              // combine halves (wave32)
                s1 += __shfl_xor(s1, 16, 32);
                if (!hi16) {                               // exclusive (m,n2) owner
                    sAgg[(m0 + 0) * 192 + col] += s0;
                    sAgg[(m0 + 1) * 192 + col] += s1;
                }
            }
        }
        __syncthreads();
    }

    for (int id = tid; id < 16 * 192; id += 256) {
        int r = id / 192, k = id % 192;
        if (i0 + r < Nq) aggG[(size_t)(b * NPAD + i0 + r) * 192 + k] = sAgg[id];
    }
}

// ---------------------------------------------------------------------------
// Node MLP: h = [agg, x] (224) -> 256 -> 256 -> 32. One wave per 16-node tile.
// A-fragments register-resident per layer; B streamed from L2-resident global.
// ---------------------------------------------------------------------------
__global__ __launch_bounds__(256)
void node_kernel(const float* __restrict__ aggG, const float* __restrict__ xsrc,
                 const _Float16* __restrict__ w0t, const _Float16* __restrict__ w1t,
                 const _Float16* __restrict__ w2t,
                 const float* __restrict__ b0, const float* __restrict__ b1,
                 const float* __restrict__ b2,
                 float* __restrict__ dst, int final_out)
{
    extern __shared__ char smem[];
    const int tid = threadIdx.x, wave = tid >> 5, lane = tid & 31;
    _Float16* slabA = (_Float16*)smem + wave * (2 * 16 * 264);
    _Float16* slab1 = slabA + 16 * 264;

    const int tile = blockIdx.x * 8 + wave;            // 0..223 (28 blocks * 8 waves)
    const int b = tile / NT, mt = tile % NT, row0 = mt * 16;
    const int lr = lane & 15, hi16 = (lane & 16) ? 1 : 0;

    // build h rows [16][224] as f16
    for (int idx = lane; idx < 16 * 224; idx += 32) {
        int r = idx / 224, k = idx % 224, node = row0 + r;
        float v = (k < 192) ? aggG[(size_t)(b * NPAD + node) * 192 + k]
                            : xsrc[(size_t)(b * NPAD + node) * 32 + (k - 192)];
        slabA[r * 264 + k] = (_Float16)v;
    }

    // ---- L0: 224 -> 256 ----
    {
        h16 Af[7];
#pragma unroll
        for (int kc = 0; kc < 7; ++kc) {
            const _Float16* ap = slabA + lr * 264 + kc * 32 + (hi16 ? 8 : 0);
            Af[kc] = ld_frag(ap, ap + 16);
        }
        for (int n = 0; n < 16; ++n) {
            f8 acc = {};
#pragma unroll
            for (int kc = 0; kc < 7; ++kc) {
                const _Float16* bp = w0t + (size_t)(n * 16 + lr) * 224 + kc * 32 + (hi16 ? 8 : 0);
                acc = WMMA_F16(Af[kc], ld_frag(bp, bp + 16), acc);
            }
            const int col = n * 16 + lr;
            const float bias = b0[col];
#pragma unroll
            for (int v = 0; v < 8; ++v)
                slab1[(v + 8 * hi16) * 264 + col] = (_Float16)lrelu(acc[v] + bias);
        }
    }

    // ---- L1: 256 -> 256 (slab1 -> slabA) ----
    {
        h16 Af[8];
#pragma unroll
        for (int kc = 0; kc < 8; ++kc) {
            const _Float16* ap = slab1 + lr * 264 + kc * 32 + (hi16 ? 8 : 0);
            Af[kc] = ld_frag(ap, ap + 16);
        }
        for (int n = 0; n < 16; ++n) {
            f8 acc = {};
#pragma unroll
            for (int kc = 0; kc < 8; ++kc) {
                const _Float16* bp = w1t + (size_t)(n * 16 + lr) * 256 + kc * 32 + (hi16 ? 8 : 0);
                acc = WMMA_F16(Af[kc], ld_frag(bp, bp + 16), acc);
            }
            const int col = n * 16 + lr;
            const float bias = b1[col];
#pragma unroll
            for (int v = 0; v < 8; ++v)
                slabA[(v + 8 * hi16) * 264 + col] = (_Float16)lrelu(acc[v] + bias);
        }
    }

    // ---- L2: 256 -> 32, no activation, f32 store ----
    {
        h16 Af[8];
#pragma unroll
        for (int kc = 0; kc < 8; ++kc) {
            const _Float16* ap = slabA + lr * 264 + kc * 32 + (hi16 ? 8 : 0);
            Af[kc] = ld_frag(ap, ap + 16);
        }
        for (int n = 0; n < 2; ++n) {
            f8 acc = {};
#pragma unroll
            for (int kc = 0; kc < 8; ++kc) {
                const _Float16* bp = w2t + (size_t)(n * 16 + lr) * 256 + kc * 32 + (hi16 ? 8 : 0);
                acc = WMMA_F16(Af[kc], ld_frag(bp, bp + 16), acc);
            }
            const int col = n * 16 + lr;
            const float bias = b2[col];
#pragma unroll
            for (int v = 0; v < 8; ++v) {
                int node = row0 + v + 8 * hi16;
                if (node < Nq) {
                    float y = acc[v] + bias;
                    if (final_out) dst[(size_t)(b * Nq + node) * 32 + col] = y;
                    else           dst[(size_t)(b * NPAD + node) * 32 + col] = y;
                }
            }
        }
    }
}

// ---------------------------------------------------------------------------
extern "C" void kernel_launch(void* const* d_in, const int* in_sizes, int n_in,
                              void* d_out, int out_size, void* d_ws, size_t ws_size,
                              hipStream_t stream)
{
    (void)in_sizes; (void)n_in; (void)out_size; (void)ws_size;
    const float* x    = (const float*)d_in[0];
    const float* few0 = (const float*)d_in[1];
    const float* feb0 = (const float*)d_in[2];
    const float* few1 = (const float*)d_in[3];
    const float* feb1 = (const float*)d_in[4];
    const float* few2 = (const float*)d_in[5];
    const float* feb2 = (const float*)d_in[6];
    const float* fnw0 = (const float*)d_in[7];
    const float* fnb0 = (const float*)d_in[8];
    const float* fnw1 = (const float*)d_in[9];
    const float* fnb1 = (const float*)d_in[10];
    const float* fnw2 = (const float*)d_in[11];
    const float* fnb2 = (const float*)d_in[12];

    char* ws = (char*)d_ws;
    size_t off = 0;
    auto alloc = [&](size_t bytes) { void* p = ws + off; off += (bytes + 255) & ~(size_t)255; return p; };

    float* xA   = (float*)alloc((size_t)Bq * NPAD * 32 * 4);
    float* xB   = (float*)alloc((size_t)Bq * NPAD * 32 * 4);
    float* P    = (float*)alloc((size_t)Bq * NPAD * 96 * 4);
    float* Qb   = (float*)alloc((size_t)Bq * NPAD * 96 * 4);
    float* aggG = (float*)alloc((size_t)Bq * NPAD * 192 * 4);
    _Float16* feW1T = (_Float16*)alloc((size_t)2 * 160 * 96 * 2);
    _Float16* feW2T = (_Float16*)alloc((size_t)2 * 192 * 160 * 2);
    _Float16* fnW0T = (_Float16*)alloc((size_t)2 * 256 * 224 * 2);
    _Float16* fnW1T = (_Float16*)alloc((size_t)2 * 256 * 256 * 2);
    _Float16* fnW2T = (_Float16*)alloc((size_t)2 * 32 * 256 * 2);

    (void)hipFuncSetAttribute(reinterpret_cast<const void*>(&edge_kernel),
                              hipFuncAttributeMaxDynamicSharedMemorySize, EDGE_LDS);
    (void)hipFuncSetAttribute(reinterpret_cast<const void*>(&node_kernel),
                              hipFuncAttributeMaxDynamicSharedMemorySize, NODE_LDS);

    init_x_kernel<<<(Bq * NPAD * 32 + 255) / 256, 256, 0, stream>>>(x, xA, xB);

    auto T = [&](_Float16* dst, const float* src, int K, int Nn) {
        int n = K * Nn;
        transpose_cvt_kernel<<<(n + 255) / 256, 256, 0, stream>>>(dst, src, K, Nn);
    };
    for (int it = 0; it < 2; ++it) {
        T(feW1T + (size_t)it * 160 * 96,  few1 + (size_t)it * 96 * 160,  96, 160);
        T(feW2T + (size_t)it * 192 * 160, few2 + (size_t)it * 160 * 192, 160, 192);
        T(fnW0T + (size_t)it * 256 * 224, fnw0 + (size_t)it * 224 * 256, 224, 256);
        T(fnW1T + (size_t)it * 256 * 256, fnw1 + (size_t)it * 256 * 256, 256, 256);
        T(fnW2T + (size_t)it * 32 * 256,  fnw2 + (size_t)it * 256 * 32,  256, 32);
    }

    for (int it = 0; it < 2; ++it) {
        const float* xsrc = (it == 0) ? xA : xB;
        pq_kernel<<<(Bq * NPAD * 96 + 255) / 256, 256, 0, stream>>>(
            xsrc, few0 + (size_t)it * 64 * 96, feb0 + (size_t)it * 96, P, Qb);

        edge_kernel<<<Bq * NT, 256, EDGE_LDS, stream>>>(
            P, Qb, feW1T + (size_t)it * 160 * 96, feW2T + (size_t)it * 192 * 160,
            feb1 + (size_t)it * 160, feb2 + (size_t)it * 192, aggG);

        node_kernel<<<28, 256, NODE_LDS, stream>>>(
            aggG, xsrc,
            fnW0T + (size_t)it * 256 * 224, fnW1T + (size_t)it * 256 * 256,
            fnW2T + (size_t)it * 32 * 256,
            fnb0 + (size_t)it * 256, fnb1 + (size_t)it * 256, fnb2 + (size_t)it * 32,
            (it == 1) ? (float*)d_out : xB, (it == 1) ? 1 : 0);
    }
}
